// Arma_83330955477199
// MI455X (gfx1250) — matrix operand order repaired and checked by reading, not compiled, not run on hardware
//
#include <hip/hip_runtime.h>
#include <stdint.h>

#define NN 50000
#define EE 320000
#define FF 256

typedef __attribute__((ext_vector_type(16))) __bf16 v16bf;
typedef __attribute__((ext_vector_type(8)))  __bf16 v8bf;
typedef __attribute__((ext_vector_type(4)))  __bf16 v4bf;
typedef __attribute__((ext_vector_type(8)))  float  v8f;

// ---------------- prep kernels ----------------

__global__ void zero_deg_kernel(float* __restrict__ deg) {
    int i = blockIdx.x * 256 + threadIdx.x;
    if (i < NN) deg[i] = 0.0f;
}

__global__ void cvt_x_kernel(const float* __restrict__ x, __bf16* __restrict__ xbf) {
    int i = blockIdx.x * 256 + threadIdx.x;
    if (i < NN * FF) xbf[i] = (__bf16)x[i];
}

// Pack W[k][f][g] (4 matrices: init0, init1, root0, root1) into the wave32
// 16-bit B-fragment layout: offset = (((m*16+gtile)*8+kb)*32+lane)*16 + j
// where element = W[f = kb*32 + (j>=8?16:0) + (lane>=16?8:0) + (j&7)][g = gtile*16 + (lane&15)]
__global__ void pack_w_kernel(const float* __restrict__ iw,
                              const float* __restrict__ rw,
                              __bf16* __restrict__ wpack) {
    int p = blockIdx.x * 256 + threadIdx.x;           // 0 .. 262143
    if (p >= 4 * 16 * 8 * 32 * 16) return;
    int j     =  p        & 15;
    int lane  = (p >> 4)  & 31;
    int kb    = (p >> 9)  & 7;
    int gtile = (p >> 12) & 15;
    int m     =  p >> 16;                             // 0..3
    int g = gtile * 16 + (lane & 15);
    int f = kb * 32 + ((j >> 3) << 4) + ((lane >> 4) << 3) + (j & 7);
    const float* W = (m < 2) ? (iw + (size_t)m * (FF * FF))
                             : (rw + (size_t)(m - 2) * (FF * FF));
    wpack[p] = (__bf16)W[(size_t)f * FF + g];
}

__global__ void deg_kernel(const long long* __restrict__ ei, float* __restrict__ deg) {
    int e = blockIdx.x * 256 + threadIdx.x;
    if (e >= EE) return;
    int d = (int)ei[EE + e];                          // dst row
    __hip_atomic_fetch_add(&deg[d], 1.0f, __ATOMIC_RELAXED, __HIP_MEMORY_SCOPE_AGENT);
}

__global__ void dinv_kernel(const float* __restrict__ deg, float* __restrict__ dinv) {
    int i = blockIdx.x * 256 + threadIdx.x;
    if (i >= NN) return;
    float d = deg[i];
    dinv[i] = (d > 0.0f) ? rsqrtf(fmaxf(d, 1.0f)) : 0.0f;
}

// ---------------- WMMA GEMM ----------------
// One wave per 16x16 output tile. wid = (rowtile*16 + gtile)*4 + m
// m in {0,1}: h_k = x @ init_w[k]  -> bf16 h buffer
// m in {2,3}: root_k = x @ root_w[k-2] + bias -> f32 agg buffer (scatter adds on top)
__global__ __launch_bounds__(256) void gemm_wmma_kernel(
        const __bf16* __restrict__ xbf,
        const __bf16* __restrict__ wpack,
        const float*  __restrict__ bias,
        __bf16* __restrict__ hbf,
        float*  __restrict__ agg) {
    const int lane    = threadIdx.x & 31;
    const int wid     = blockIdx.x * 8 + (threadIdx.x >> 5);   // 0..199999
    const int m       =  wid       & 3;
    const int gtile   = (wid >> 2) & 15;
    const int rowtile =  wid >> 6;                             // 0..3124
    const int half    = lane >> 4;
    const int mrow    = lane & 15;

    // A: row = rowtile*16 + (lane&15); per K-block of 32 the lane holds
    // 8 values at kb*32 + half*8 and 8 at kb*32 + 16 + half*8 (ISA 16-bit A layout)
    const __bf16* arow = xbf + ((size_t)(rowtile * 16 + mrow)) * FF + half * 8;
    // B: pre-packed contiguous 16 bf16 per lane per K-block
    const __bf16* bptr = wpack + ((size_t)((m * 16 + gtile) * 8) * 32 + lane) * 16;

    v8f c = {};
#pragma unroll
    for (int kb = 0; kb < 8; ++kb) {
        v8bf a0 = *(const v8bf*)(arow + kb * 32);
        v8bf a1 = *(const v8bf*)(arow + kb * 32 + 16);
        v8bf b0 = *(const v8bf*)(bptr + (size_t)kb * (32 * 16));
        v8bf b1 = *(const v8bf*)(bptr + (size_t)kb * (32 * 16) + 8);
        v16bf A, B;
#pragma unroll
        for (int i = 0; i < 8; ++i) {
            A[i] = a0[i]; A[i + 8] = a1[i];
            B[i] = b0[i]; B[i + 8] = b1[i];
        }
        c = __builtin_amdgcn_wmma_f32_16x16x32_bf16(
                false, A, false, B, (short)0, c, false, false);
    }

    // D layout: lanes 0-15 VGPR r -> (M=r, N=lane); lanes 16-31 -> (M=r+8, N=lane-16)
    const int col   = gtile * 16 + (lane & 15);
    const int rbase = rowtile * 16 + half * 8;
    if (m >= 2) {
        const int k = m - 2;
        const float bv = bias[k * FF + col];
        float* aout = agg + ((size_t)k * NN + rbase) * FF + col;
#pragma unroll
        for (int r = 0; r < 8; ++r)
            aout[(size_t)r * FF] = c[r] + bv;
    } else {
        __bf16* hout = hbf + ((size_t)m * NN + rbase) * FF + col;
#pragma unroll
        for (int r = 0; r < 8; ++r)
            hout[(size_t)r * FF] = (__bf16)c[r];
    }
}

// ---------------- edge scatter ----------------
// 64 threads per edge, 4 features each, both K stacks.
__global__ void scatter_kernel(const long long* __restrict__ ei,
                               const float* __restrict__ dinv,
                               const __bf16* __restrict__ hbf,
                               float* __restrict__ agg) {
    int t = blockIdx.x * 256 + threadIdx.x;           // 0 .. EE*64-1
    int e = t >> 6;
    int c = (t & 63) << 2;
    if (e >= EE) return;
    int s = (int)ei[e];
    int d = (int)ei[EE + e];
    float w = dinv[s] * dinv[d];
    if (w == 0.0f) return;
#pragma unroll
    for (int k = 0; k < 2; ++k) {
        const __bf16* hp = hbf + ((size_t)k * NN + s) * FF + c;
        float*        ap = agg + ((size_t)k * NN + d) * FF + c;
        v4bf hv = *(const v4bf*)hp;
#pragma unroll
        for (int i = 0; i < 4; ++i)
            __hip_atomic_fetch_add(&ap[i], w * (float)hv[i],
                                   __ATOMIC_RELAXED, __HIP_MEMORY_SCOPE_AGENT);
    }
}

// ---------------- epilogue ----------------
__global__ void final_kernel(const float* __restrict__ x,
                             const float* __restrict__ agg,
                             float* __restrict__ out) {
    int i = blockIdx.x * 256 + threadIdx.x;
    if (i >= NN * FF) return;
    float a0 = agg[i];
    float a1 = agg[(size_t)NN * FF + i];
    float arma = 0.5f * (fmaxf(a0, 0.0f) + fmaxf(a1, 0.0f));
    out[i] = x[i] + fmaxf(arma, 0.0f);
}

// ---------------- launcher ----------------
extern "C" void kernel_launch(void* const* d_in, const int* in_sizes, int n_in,
                              void* d_out, int out_size, void* d_ws, size_t ws_size,
                              hipStream_t stream) {
    const float*     x  = (const float*)d_in[0];
    const long long* ei = (const long long*)d_in[1];   // int64 [2,E]
    const float*     iw = (const float*)d_in[2];       // [K,F,F]
    const float*     rw = (const float*)d_in[3];       // [K,F,F]
    const float*     bs = (const float*)d_in[4];       // [K,1,F]
    float* out = (float*)d_out;

    auto alignup = [](size_t v) { return (v + 255) & ~(size_t)255; };
    char* ws = (char*)d_ws;
    size_t o = 0;
    float*  deg   = (float*)(ws + o);  o = alignup(o + (size_t)NN * 4);
    float*  dinv  = (float*)(ws + o);  o = alignup(o + (size_t)NN * 4);
    __bf16* xbf   = (__bf16*)(ws + o); o = alignup(o + (size_t)NN * FF * 2);
    __bf16* wpack = (__bf16*)(ws + o); o = alignup(o + (size_t)4 * 16 * 8 * 32 * 16 * 2);
    __bf16* hbf   = (__bf16*)(ws + o); o = alignup(o + (size_t)2 * NN * FF * 2);
    float*  agg   = (float*)(ws + o);  o = alignup(o + (size_t)2 * NN * FF * 4);

    zero_deg_kernel<<<(NN + 255) / 256, 256, 0, stream>>>(deg);
    cvt_x_kernel<<<(NN * FF + 255) / 256, 256, 0, stream>>>(x, xbf);
    pack_w_kernel<<<(4 * 16 * 8 * 32 * 16) / 256, 256, 0, stream>>>(iw, rw, wpack);
    deg_kernel<<<(EE + 255) / 256, 256, 0, stream>>>(ei, deg);
    dinv_kernel<<<(NN + 255) / 256, 256, 0, stream>>>(deg, dinv);
    // 3125 row tiles * 16 gtiles * 4 matrices = 200000 waves / 8 per block
    gemm_wmma_kernel<<<25000, 256, 0, stream>>>(xbf, wpack, bs, hbf, agg);
    scatter_kernel<<<(EE * 64) / 256, 256, 0, stream>>>(ei, dinv, hbf, agg);
    final_kernel<<<(NN * FF + 255) / 256, 256, 0, stream>>>(x, agg, out);
}
